// mambalayer_90331752169739
// MI455X (gfx1250) — matrix operand order repaired and checked by reading, not compiled, not run on hardware
//
#include <hip/hip_runtime.h>
#include <hip/hip_bf16.h>

// ---------------------------------------------------------------------------
// Mamba layer for MI455X (gfx1250): f16 WMMA GEMMs (f32 accum), LDS-staged
// depthwise conv, wave-per-channel selective scan with shuffle reductions.
// All GEMM N dims padded to multiples of 64 -> branch-free WMMA inner loop.
// ---------------------------------------------------------------------------

#define D_MODEL 640
#define D_STATE 64
#define D_CONV  640
#define DT_RANK 40
#define SEQ_L   10240
#define XDBL_N  192   // 168 padded to 12*16 (divisible by 64 -> no tile guards)
#define DT_K    64    // 40 padded to 2*32

typedef __attribute__((ext_vector_type(16))) _Float16 v16h;
typedef __attribute__((ext_vector_type(8)))  _Float16 v8h;
typedef __attribute__((ext_vector_type(8)))  float    v8f;

// ----------------------------- WMMA fragment loaders -----------------------
// A (16x32 f16, M x K): lane L holds row M = L&15.
//   lanes 0-15:  elems 0..15 = K 0..15
//   lanes 16-31: elems 0..15 = K 16..31
__device__ inline v16h load_a_frag(const _Float16* __restrict__ p) {
  v8h lo = *(const v8h*)p;         // K = k0 + 16*half + [0..7]
  v8h hi = *(const v8h*)(p + 8);   // K = k0 + 16*half + 8 + [0..7]
  return __builtin_shufflevector(lo, hi, 0,1,2,3,4,5,6,7,8,9,10,11,12,13,14,15);
}

// B (32x16 f16, K x N): lane L holds column N = L&15;
//   lanes 0-15 carry K = k0..k0+15, lanes 16-31 carry K = k0+16..k0+31.
// Weights are row-major [N, K] (out = act @ W.T) -> one contiguous 32B read.
__device__ inline v16h load_b_frag(const _Float16* __restrict__ p) {
  v8h lo = *(const v8h*)p;
  v8h hi = *(const v8h*)(p + 8);
  return __builtin_shufflevector(lo, hi, 0,1,2,3,4,5,6,7,8,9,10,11,12,13,14,15);
}

// ----------------------------- GEMM: C[M,ldc] = A[M,K] @ W[N,K]^T ----------
// One wave -> 16x64 output strip (4 wmma tiles), branch-free inner loop.
// 8 waves/block = 8 consecutive row tiles. Grid: (N/64, M/128). N % 64 == 0.
__global__ __launch_bounds__(256)
void gemm_wmma_f16(const _Float16* __restrict__ A, int lda,
                   const _Float16* __restrict__ W, int ldw,
                   float* __restrict__ C, int ldc, int K) {
  const int lane = threadIdx.x & 31;
  const int wave = threadIdx.x >> 5;
  const int m0 = (blockIdx.y * 8 + wave) * 16;
  const int n0 = blockIdx.x * 64;
  const int half = lane >> 4;
  const int l15  = lane & 15;

  // Per-lane base pointers (advance by 32 halves per k-step)
  const _Float16* pa  = A + (long)(m0 + l15) * lda + 16 * half;
  const _Float16* pb0 = W + (long)(n0 +  0 + l15) * ldw + 16 * half;
  const _Float16* pb1 = W + (long)(n0 + 16 + l15) * ldw + 16 * half;
  const _Float16* pb2 = W + (long)(n0 + 32 + l15) * ldw + 16 * half;
  const _Float16* pb3 = W + (long)(n0 + 48 + l15) * ldw + 16 * half;

  v8f acc0 = {}, acc1 = {}, acc2 = {}, acc3 = {};
  for (int k0 = 0; k0 < K; k0 += 32) {
    v16h a  = load_a_frag(pa + k0);
    v16h b0 = load_b_frag(pb0 + k0);
    v16h b1 = load_b_frag(pb1 + k0);
    v16h b2 = load_b_frag(pb2 + k0);
    v16h b3 = load_b_frag(pb3 + k0);
    acc0 = __builtin_amdgcn_wmma_f32_16x16x32_f16(false, a, false, b0, (short)0, acc0, false, false);
    acc1 = __builtin_amdgcn_wmma_f32_16x16x32_f16(false, a, false, b1, (short)0, acc1, false, false);
    acc2 = __builtin_amdgcn_wmma_f32_16x16x32_f16(false, a, false, b2, (short)0, acc2, false, false);
    acc3 = __builtin_amdgcn_wmma_f32_16x16x32_f16(false, a, false, b3, (short)0, acc3, false, false);
  }

  // C/D layout: VGPR r -> M = r (lanes 0-15) or r+8 (lanes 16-31); N = lane&15
  float* crow = C + (long)(m0 + 8 * half) * ldc + n0 + l15;
#pragma unroll
  for (int r = 0; r < 8; ++r) {
    float* c = crow + (long)r * ldc;
    c[0]  = acc0[r];
    c[16] = acc1[r];
    c[32] = acc2[r];
    c[48] = acc3[r];
  }
}

// ----------------------------- gather + LayerNorm --> f16 ------------------
// flat[l, c] = coords[(l%32)*640*320 + c*320 + (l/32)], LN over c (640).
__global__ __launch_bounds__(256)
void gather_ln_kernel(const float* __restrict__ coords,
                      const float* __restrict__ ln_w,
                      const float* __restrict__ ln_b,
                      _Float16* __restrict__ xln_h) {
  const int l = blockIdx.x;
  const int w = l & 31, s = l >> 5;
  const float* base = coords + (long)w * (D_MODEL * 320) + s;
  __shared__ float red[256];

  float v[3];
  float sum = 0.f;
  for (int c = threadIdx.x, i = 0; c < D_MODEL; c += 256, ++i) {
    float x = base[(long)c * 320];
    v[i] = x;
    sum += x;
  }
  red[threadIdx.x] = sum; __syncthreads();
  for (int o = 128; o > 0; o >>= 1) {
    if (threadIdx.x < o) red[threadIdx.x] += red[threadIdx.x + o];
    __syncthreads();
  }
  const float mu = red[0] * (1.f / D_MODEL);
  __syncthreads();

  float sq = 0.f;
  for (int c = threadIdx.x, i = 0; c < D_MODEL; c += 256, ++i) {
    float d = v[i] - mu; sq += d * d;
  }
  red[threadIdx.x] = sq; __syncthreads();
  for (int o = 128; o > 0; o >>= 1) {
    if (threadIdx.x < o) red[threadIdx.x] += red[threadIdx.x + o];
    __syncthreads();
  }
  const float rstd = rsqrtf(red[0] * (1.f / D_MODEL) + 1e-5f);

  for (int c = threadIdx.x, i = 0; c < D_MODEL; c += 256, ++i) {
    float y = (v[i] - mu) * rstd * ln_w[c] + ln_b[c];
    xln_h[(long)l * D_MODEL + c] = (_Float16)y;
  }
}

// ----------------------------- f32 -> f16 with zero padding ----------------
__global__ __launch_bounds__(256)
void cvt_pad_f16(const float* __restrict__ src, int srcStride, int srcRows,
                 int srcCols, _Float16* __restrict__ dst, int dstCols,
                 long total) {
  long i = (long)blockIdx.x * 256 + threadIdx.x;
  if (i >= total) return;
  int r = (int)(i / dstCols), c = (int)(i % dstCols);
  float v = (r < srcRows && c < srcCols) ? src[(long)r * srcStride + c] : 0.f;
  dst[i] = (_Float16)v;
}

// ----------------------------- transpose xz[:, :640] -> xt[640, L] ---------
__global__ __launch_bounds__(256)
void transpose_x_kernel(const float* __restrict__ xz, float* __restrict__ xt) {
  __shared__ float tile[32][33];
  const int c0 = blockIdx.x * 32;
  const int l0 = blockIdx.y * 32;
  const int tx = threadIdx.x, ty = threadIdx.y;   // block (32, 8)
#pragma unroll
  for (int i = 0; i < 32; i += 8)
    tile[ty + i][tx] = xz[(long)(l0 + ty + i) * (2 * D_MODEL) + c0 + tx];
  __syncthreads();
#pragma unroll
  for (int i = 0; i < 32; i += 8)
    xt[(long)(c0 + ty + i) * SEQ_L + l0 + tx] = tile[tx][ty + i];
}

// ----------------------------- depthwise causal conv (k=640) + SiLU --------
// One block = one channel x 256 outputs. Weights + input segment in LDS.
__global__ __launch_bounds__(256)
void conv_silu_kernel(const float* __restrict__ xt,      // [640, L] channel-major
                      const float* __restrict__ conv_w,  // [640, 640]
                      const float* __restrict__ conv_b,
                      float* __restrict__ x_silu,        // [L, 640]
                      _Float16* __restrict__ x_silu_h) { // [L, 640]
  const int c = blockIdx.y;
  const int l0 = blockIdx.x * 256;
  __shared__ float wbuf[D_CONV];
  __shared__ float seg[256 + D_CONV - 1];   // 895

  for (int i = threadIdx.x; i < D_CONV; i += 256)
    wbuf[i] = conv_w[(long)c * D_CONV + i];
  for (int i = threadIdx.x; i < 256 + D_CONV - 1; i += 256) {
    int l = l0 - (D_CONV - 1) + i;
    seg[i] = (l >= 0) ? xt[(long)c * SEQ_L + l] : 0.f;
  }
  __syncthreads();

  float acc = conv_b[c];
#pragma unroll 8
  for (int t = 0; t < D_CONV; ++t)
    acc = fmaf(seg[threadIdx.x + t], wbuf[t], acc);

  const float sig = 1.f / (1.f + __expf(-acc));
  const float v = acc * sig;
  const long l = l0 + threadIdx.x;
  x_silu[l * D_MODEL + c] = v;
  x_silu_h[l * D_MODEL + c] = (_Float16)v;
}

// ----------------------------- dt = softplus(dt_lin + bias), in place ------
__global__ __launch_bounds__(256)
void dt_softplus_kernel(float* __restrict__ dt, const float* __restrict__ bias) {
  long i = (long)blockIdx.x * 256 + threadIdx.x;
  if (i >= (long)SEQ_L * D_MODEL) return;
  float x = dt[i] + bias[(int)(i % D_MODEL)];
  dt[i] = (x > 20.f) ? x : log1pf(__expf(x));
}

// ----------------------------- selective scan ------------------------------
// One wave32 per channel d; lane j owns states n=j and n=j+32. Serial over L,
// wave shuffle-reduce per step for y[t,d] = sum_n h*c.
__global__ __launch_bounds__(256)
void scan_kernel(const float* __restrict__ dt,     // [L, 640]
                 const float* __restrict__ x_silu, // [L, 640]
                 const float* __restrict__ xdbl,   // [L, 192]: dt|B(40..103)|C(104..167)
                 const float* __restrict__ A_log,  // [640, 64]
                 float* __restrict__ ys) {         // [L, 640]
  const int lane = threadIdx.x & 31;
  const int wave = threadIdx.x >> 5;
  const int d = blockIdx.x * 8 + wave;

  const float a0 = -__expf(A_log[(long)d * D_STATE + lane]);
  const float a1 = -__expf(A_log[(long)d * D_STATE + 32 + lane]);
  float h0 = 0.f, h1 = 0.f;

  for (int t = 0; t < SEQ_L; ++t) {
    const float* row = xdbl + (long)t * XDBL_N;
    const float dtt = dt[(long)t * D_MODEL + d];
    const float u   = x_silu[(long)t * D_MODEL + d];
    const float b0 = row[40 + lane],  b1 = row[72 + lane];
    const float c0 = row[104 + lane], c1 = row[136 + lane];
    const float du = dtt * u;
    h0 = __expf(dtt * a0) * h0 + du * b0;
    h1 = __expf(dtt * a1) * h1 + du * b1;
    float p = h0 * c0 + h1 * c1;
#pragma unroll
    for (int off = 16; off > 0; off >>= 1)
      p += __shfl_xor(p, off, 32);
    if (lane == 0) ys[(long)t * D_MODEL + d] = p;
  }
}

// ----------------------------- y = (ys + x*D) * silu(z) -> f16 -------------
__global__ __launch_bounds__(256)
void gate_kernel(const float* __restrict__ ys, const float* __restrict__ x_silu,
                 const float* __restrict__ xz, const float* __restrict__ Dv,
                 _Float16* __restrict__ gated_h) {
  long i = (long)blockIdx.x * 256 + threadIdx.x;
  if (i >= (long)SEQ_L * D_MODEL) return;
  int l = (int)(i / D_MODEL), d = (int)(i % D_MODEL);
  float y = ys[i] + x_silu[i] * Dv[d];
  float z = xz[(long)l * (2 * D_MODEL) + D_MODEL + d];
  float sz = z / (1.f + __expf(-z));
  gated_h[i] = (_Float16)(y * sz);
}

// ----------------------------- final permute + add -------------------------
// d_out = [ output(16,640,320) | coords_out(1,32,640,320) ]
__global__ __launch_bounds__(256)
void finalize_kernel(const float* __restrict__ out, float* __restrict__ dout) {
  long idx = (long)blockIdx.x * 256 + threadIdx.x;  // over 32*640*320
  if (idx >= (long)32 * D_MODEL * 320) return;
  int s = (int)(idx % 320);
  int c = (int)((idx / 320) % D_MODEL);
  int w = (int)(idx / (320 * D_MODEL));
  long l = (long)s * 32 + w;
  float v = out[l * D_MODEL + c];
  dout[(long)16 * D_MODEL * 320 + idx] = v;          // coords_out
  if (w < 16) {
    long l2 = (long)s * 32 + w + 16;
    dout[idx] = v + out[l2 * D_MODEL + c];           // output = c0[:16]+c0[16:]
  }
}

// ---------------------------------------------------------------------------
extern "C" void kernel_launch(void* const* d_in, const int* in_sizes, int n_in,
                              void* d_out, int out_size, void* d_ws, size_t ws_size,
                              hipStream_t stream) {
  const float* coords    = (const float*)d_in[0];
  const float* ln_w      = (const float*)d_in[1];
  const float* ln_b      = (const float*)d_in[2];
  const float* in_proj_w = (const float*)d_in[3];   // [1280, 640]
  const float* conv_w    = (const float*)d_in[4];   // [640, 1, 640]
  const float* conv_b    = (const float*)d_in[5];
  const float* x_proj_w  = (const float*)d_in[6];   // [168, 640]
  const float* dt_proj_w = (const float*)d_in[7];   // [640, 40]
  const float* dt_proj_b = (const float*)d_in[8];
  const float* A_log     = (const float*)d_in[9];   // [640, 64]
  const float* Dv        = (const float*)d_in[10];
  const float* out_proj_w= (const float*)d_in[11];  // [640, 640]

  char* p = (char*)d_ws;
  auto alloc = [&](size_t bytes) -> void* {
    void* r = (void*)p;
    p += (bytes + 255) & ~(size_t)255;
    return r;
  };
  const long L = SEQ_L;
  _Float16* xln_h    = (_Float16*)alloc(L * D_MODEL * 2);
  _Float16* w_in_h   = (_Float16*)alloc((size_t)2 * D_MODEL * D_MODEL * 2);
  _Float16* w_xp_h   = (_Float16*)alloc((size_t)XDBL_N * D_MODEL * 2);
  _Float16* w_dt_h   = (_Float16*)alloc((size_t)D_MODEL * DT_K * 2);
  _Float16* w_out_h  = (_Float16*)alloc((size_t)D_MODEL * D_MODEL * 2);
  float*    xz       = (float*)alloc(L * 2 * D_MODEL * 4);
  float*    xt       = (float*)alloc((size_t)D_MODEL * L * 4);
  float*    x_silu   = (float*)alloc(L * D_MODEL * 4);
  _Float16* x_silu_h = (_Float16*)alloc(L * D_MODEL * 2);
  float*    xdbl     = (float*)alloc(L * XDBL_N * 4);
  _Float16* dt_h     = (_Float16*)alloc(L * DT_K * 2);
  float*    dt_buf   = (float*)alloc(L * D_MODEL * 4);
  float*    ys       = (float*)alloc(L * D_MODEL * 4);
  _Float16* gated_h  = (_Float16*)alloc(L * D_MODEL * 2);
  float*    outbuf   = (float*)alloc(L * D_MODEL * 4);

  // --- stage weights to f16 (zero-padded so every GEMM N % 64 == 0, K % 32 == 0)
  {
    long t1 = (long)2 * D_MODEL * D_MODEL;
    cvt_pad_f16<<<(t1 + 255) / 256, 256, 0, stream>>>(
        in_proj_w, D_MODEL, 2 * D_MODEL, D_MODEL, w_in_h, D_MODEL, t1);
    long t2 = (long)XDBL_N * D_MODEL;   // rows 168..191 zero
    cvt_pad_f16<<<(t2 + 255) / 256, 256, 0, stream>>>(
        x_proj_w, D_MODEL, DT_RANK + 2 * D_STATE, D_MODEL, w_xp_h, D_MODEL, t2);
    long t3 = (long)D_MODEL * DT_K;     // cols 40..63 zero
    cvt_pad_f16<<<(t3 + 255) / 256, 256, 0, stream>>>(
        dt_proj_w, DT_RANK, D_MODEL, DT_RANK, w_dt_h, DT_K, t3);
    long t4 = (long)D_MODEL * D_MODEL;
    cvt_pad_f16<<<(t4 + 255) / 256, 256, 0, stream>>>(
        out_proj_w, D_MODEL, D_MODEL, D_MODEL, w_out_h, D_MODEL, t4);
  }

  // --- gather + layernorm -> xln_h [L, 640] f16
  gather_ln_kernel<<<SEQ_L, 256, 0, stream>>>(coords, ln_w, ln_b, xln_h);

  // --- in_proj: xz[L,1280] = xln @ in_proj_w.T   (WMMA, N=1280)
  gemm_wmma_f16<<<dim3(2 * D_MODEL / 64, SEQ_L / 128), 256, 0, stream>>>(
      xln_h, D_MODEL, w_in_h, D_MODEL, xz, 2 * D_MODEL, D_MODEL);

  // --- transpose x half of xz to channel-major for the conv
  transpose_x_kernel<<<dim3(D_MODEL / 32, SEQ_L / 32), dim3(32, 8), 0, stream>>>(xz, xt);

  // --- depthwise causal conv (k=640) + bias + SiLU
  conv_silu_kernel<<<dim3(SEQ_L / 256, D_MODEL), 256, 0, stream>>>(
      xt, conv_w, conv_b, x_silu, x_silu_h);

  // --- x_proj: xdbl[L,192] = x_silu @ x_proj_w.T   (WMMA, N padded 168->192)
  gemm_wmma_f16<<<dim3(XDBL_N / 64, SEQ_L / 128), 256, 0, stream>>>(
      x_silu_h, D_MODEL, w_xp_h, D_MODEL, xdbl, XDBL_N, D_MODEL);

  // --- dt_raw (cols 0..39 of xdbl) -> f16, K padded 40->64
  {
    long t = L * DT_K;
    cvt_pad_f16<<<(t + 255) / 256, 256, 0, stream>>>(
        xdbl, XDBL_N, SEQ_L, DT_RANK, dt_h, DT_K, t);
  }

  // --- dt_proj: dt_buf[L,640] = dt_raw @ dt_proj_w.T   (WMMA, K=64)
  gemm_wmma_f16<<<dim3(D_MODEL / 64, SEQ_L / 128), 256, 0, stream>>>(
      dt_h, DT_K, w_dt_h, DT_K, dt_buf, D_MODEL, DT_K);

  // --- softplus(dt + bias)
  {
    long t = L * D_MODEL;
    dt_softplus_kernel<<<(t + 255) / 256, 256, 0, stream>>>(dt_buf, dt_proj_b);
  }

  // --- selective scan: 640 waves, one per channel
  scan_kernel<<<D_MODEL / 8, 256, 0, stream>>>(dt_buf, x_silu, xdbl, A_log, ys);

  // --- gating: (ys + x*D) * silu(z) -> f16
  {
    long t = L * D_MODEL;
    gate_kernel<<<(t + 255) / 256, 256, 0, stream>>>(ys, x_silu, xz, Dv, gated_h);
  }

  // --- out_proj: out[L,640] = gated @ out_proj_w.T   (WMMA)
  gemm_wmma_f16<<<dim3(D_MODEL / 64, SEQ_L / 128), 256, 0, stream>>>(
      gated_h, D_MODEL, w_out_h, D_MODEL, outbuf, D_MODEL, D_MODEL);

  // --- permute back, write (output, coords_out) into d_out
  {
    long t = (long)32 * D_MODEL * 320;
    finalize_kernel<<<(t + 255) / 256, 256, 0, stream>>>(outbuf, (float*)d_out);
  }
}
